// LazyDecoderBlock_4612794876262
// MI455X (gfx1250) — matrix operand order repaired and checked by reading, not compiled, not run on hardware
//
#include <hip/hip_runtime.h>
#include <hip/hip_bf16.h>
#include <math.h>

// ---------------------------------------------------------------------------
// Problem constants (match the JAX reference)
// ---------------------------------------------------------------------------
#define D_MODEL 2048
#define N_HEADS 16
#define GKV     4
#define D_FF    8192
#define D_HEAD  128          // D_MODEL / N_HEADS
#define BATCH   2
#define SEQ     2048
#define EPS     1e-6f

typedef __bf16 bf16;
typedef __attribute__((ext_vector_type(16))) __bf16 bf16x16;   // 8 VGPRs
typedef __attribute__((ext_vector_type(8)))  float  floatx8;   // 8 VGPRs

union FragBF {
    bf16x16      v;
    unsigned int u[8];
};

__device__ __forceinline__ bf16 f2bf(float x) { return (bf16)x; }

// ---------------------------------------------------------------------------
// Async global->LDS staging (CDNA5 ASYNCcnt path) with synchronous fallback.
// The builtin takes 16-byte int-vector pointers: AS1 source, AS3 destination.
// ---------------------------------------------------------------------------
#if __has_builtin(__builtin_amdgcn_global_load_async_to_lds_b128) && \
    __has_builtin(__builtin_amdgcn_s_wait_asynccnt)
#define USE_ASYNC_COPY 1
#else
#define USE_ASYNC_COPY 0
#endif

typedef int v4i_vec __attribute__((vector_size(16)));
#define AS_GLOBAL __attribute__((address_space(1)))
#define AS_SHARED __attribute__((address_space(3)))

__device__ __forceinline__ void g2l_b128(const bf16* g, bf16* l) {
#if USE_ASYNC_COPY
    __builtin_amdgcn_global_load_async_to_lds_b128(
        (AS_GLOBAL v4i_vec*)g, (AS_SHARED v4i_vec*)l, 0, 0);
#else
    *(uint4*)l = *(const uint4*)g;
#endif
}
__device__ __forceinline__ void wait_async_prev_stage() {
#if USE_ASYNC_COPY
    __builtin_amdgcn_s_wait_asynccnt(4);   // 4 async ops in flight per stage/wave
#endif
}
__device__ __forceinline__ void wait_async_all() {
#if USE_ASYNC_COPY
    __builtin_amdgcn_s_wait_asynccnt(0);
#endif
}

// ---------------------------------------------------------------------------
// f32 -> bf16 conversion (weights / kv context), grid-stride
// ---------------------------------------------------------------------------
__global__ void k_f32_to_bf16(const float* __restrict__ in,
                              bf16* __restrict__ out, long n) {
    long i = (long)blockIdx.x * blockDim.x + threadIdx.x;
    long stride = (long)gridDim.x * blockDim.x;
    for (; i < n; i += stride) out[i] = f2bf(in[i]);
}

// ---------------------------------------------------------------------------
// RMSNorm over D_MODEL, one block per row, bf16 output (feeds WMMA GEMMs)
// ---------------------------------------------------------------------------
__global__ __launch_bounds__(256)
void k_rmsnorm_bf16(const float* __restrict__ x, const float* __restrict__ w,
                    bf16* __restrict__ out) {
    __shared__ float red[256];
    const long row = blockIdx.x;
    const float* xr = x + row * (long)D_MODEL;
    float s = 0.f;
    for (int i = threadIdx.x; i < D_MODEL; i += 256) {
        float v = xr[i];
        s += v * v;
    }
    red[threadIdx.x] = s;
    __syncthreads();
    for (int off = 128; off > 0; off >>= 1) {
        if ((int)threadIdx.x < off) red[threadIdx.x] += red[threadIdx.x + off];
        __syncthreads();
    }
    const float inv = rsqrtf(red[0] / (float)D_MODEL + EPS);
    bf16* orow = out + row * (long)D_MODEL;
    for (int i = threadIdx.x; i < D_MODEL; i += 256)
        orow[i] = f2bf(xr[i] * inv * w[i]);
}

// ---------------------------------------------------------------------------
// bf16 WMMA GEMM:  D[M,N] = act( A[M,K] @ Bw[K,N] + Res[M,N] )
// Block tile 128x128, K-step 32, double-buffered async global->LDS staging.
// 8 waves: 4 (M) x 2 (N); each wave owns 2x4 v_wmma_f32_16x16x32_bf16 frags.
// Template flags keep the epilogue branch-free.
// All dims are multiples of 128 in this workload -> no bounds checks.
// ---------------------------------------------------------------------------
template <bool HAS_RES, bool ACT_SILU, bool OUT_BF16>
__global__ __launch_bounds__(256)
void k_gemm_bf16_wmma(const bf16* __restrict__ A, const bf16* __restrict__ Bw,
                      const float* Res, float* Df, bf16* Dbf,
                      int M, int N, int K) {
    __shared__ alignas(16) bf16 As[2][128][32];   // [buf][m][k]
    __shared__ alignas(16) bf16 Bs[2][32][128];   // [buf][k][n]

    const int tid  = threadIdx.x;
    const int lane = tid & 31;
    const int wave = tid >> 5;
    const int m0 = blockIdx.y * 128;
    const int n0 = blockIdx.x * 128;
    const int wm = wave & 3;            // M subtile: 32 rows
    const int wn = wave >> 2;           // N subtile: 64 cols
    const int hi  = (lane >= 16) ? 1 : 0;
    const int l16 = lane & 15;

    // Stage = 1024 x 16B transfers (A: 512, B: 512) -> 4 per thread,
    // i.e. exactly 4 async instructions per wave per stage.
    const int ar0 = tid >> 2;            // A rows: ar0 and ar0+64
    const int as0 = (tid & 3) * 8;       // A k segment (8 bf16 = 16B)
    const int br0 = tid >> 4;            // B rows: br0 and br0+16
    const int bs0 = (tid & 15) * 8;      // B n segment

    auto stage = [&](int buf, int k0) {
        g2l_b128(A  + (size_t)(m0 + ar0)      * K + k0 + as0, &As[buf][ar0][as0]);
        g2l_b128(A  + (size_t)(m0 + ar0 + 64) * K + k0 + as0, &As[buf][ar0 + 64][as0]);
        g2l_b128(Bw + (size_t)(k0 + br0)      * N + n0 + bs0, &Bs[buf][br0][bs0]);
        g2l_b128(Bw + (size_t)(k0 + br0 + 16) * N + n0 + bs0, &Bs[buf][br0 + 16][bs0]);
    };

    floatx8 acc[2][4] = {};
    const int nk = K >> 5;
    stage(0, 0);

    for (int kt = 0; kt < nk; ++kt) {
        const int buf = kt & 1;
        if (kt + 1 < nk) {
            stage(buf ^ 1, (kt + 1) * 32);   // prefetch next tile via DMA
            wait_async_prev_stage();         // current tile's 4 ops retired
        } else {
            wait_async_all();
        }
        __syncthreads();                     // publish current tile

        // A fragments (ISA 7.12.2 16-bit A 16x32 layout)
        FragBF afr[2];
#pragma unroll
        for (int am = 0; am < 2; ++am) {
            const int row = wm * 32 + am * 16 + l16;
#pragma unroll
            for (int j = 0; j < 8; ++j) {
                const int koff = 2 * (j & 3) + ((j >= 4) ? 16 : 0) + hi * 8;
                afr[am].u[j] = *(const unsigned int*)&As[buf][row][koff];
            }
        }
        // B fragments (32x16: K striped across lane halves, n = lane%16)
        FragBF bfr[4];
#pragma unroll
        for (int bn = 0; bn < 4; ++bn) {
            const int col = wn * 64 + bn * 16 + l16;
#pragma unroll
            for (int v = 0; v < 8; ++v) {
                const int kk = 2 * (v & 3) + ((v >= 4) ? 16 : 0) + hi * 8;
                const unsigned short lo = *(const unsigned short*)&Bs[buf][kk][col];
                const unsigned short hh = *(const unsigned short*)&Bs[buf][kk + 1][col];
                bfr[bn].u[v] = (unsigned int)lo | ((unsigned int)hh << 16);
            }
        }
#pragma unroll
        for (int am = 0; am < 2; ++am)
#pragma unroll
            for (int bn = 0; bn < 4; ++bn)
                acc[am][bn] = __builtin_amdgcn_wmma_f32_16x16x32_bf16(
                    false, afr[am].v, false, bfr[bn].v,
                    (short)0, acc[am][bn], false, false);

        __syncthreads();                     // all reads done before buf reuse
    }

    // Epilogue (C/D layout: lane -> n = lane%16, vgpr r -> m = r + 8*hi)
#pragma unroll
    for (int am = 0; am < 2; ++am) {
#pragma unroll
        for (int bn = 0; bn < 4; ++bn) {
            const int col = n0 + wn * 64 + bn * 16 + l16;
#pragma unroll
            for (int r = 0; r < 8; ++r) {
                const int row = m0 + wm * 32 + am * 16 + r + hi * 8;
                const size_t idx = (size_t)row * N + col;
                float v = acc[am][bn][r];
                if (HAS_RES)  v += Res[idx];
                if (ACT_SILU) v = v * __builtin_amdgcn_rcpf(1.f + __expf(-v));
                if (OUT_BF16) Dbf[idx] = f2bf(v);
                else          Df[idx]  = v;
            }
        }
    }
}

// ---------------------------------------------------------------------------
// Fused flash attention, one wave (32 threads) per (batch, head, 16-q tile).
// Keys processed in tiles of 32 so P@V uses WMMA's K=32 exactly.
// Q/K/V bf16, f32 accumulation, online softmax. kv_ratio: 1 self, 4 cross GQA.
// ---------------------------------------------------------------------------
__global__ __launch_bounds__(32)
void k_flash_attn(const bf16* __restrict__ Q, long q_row_stride,
                  const bf16* __restrict__ Kp, const bf16* __restrict__ Vp,
                  long kv_row_stride, int kv_ratio,
                  bf16* __restrict__ O, long o_row_stride,
                  int T, int causal, float scale) {
    const int lane = threadIdx.x;
    const int qt = blockIdx.x, h = blockIdx.y, b = blockIdx.z;
    const int qi0 = qt * 16;
    const int hi  = (lane >= 16) ? 1 : 0;
    const int l16 = lane & 15;

    const bf16* qbase = Q  + (size_t)b * T * q_row_stride  + (size_t)h * D_HEAD;
    const bf16* kbase = Kp + (size_t)b * T * kv_row_stride + (size_t)(h / kv_ratio) * D_HEAD;
    const bf16* vbase = Vp + (size_t)b * T * kv_row_stride + (size_t)(h / kv_ratio) * D_HEAD;

    __shared__ alignas(8) float Ssm[16][32];
    __shared__ alignas(8) bf16  Psm[16][32];
    __shared__ float csm[16];
    __shared__ float lsm[16];

    // Q fragments: 16 rows x 128 (4 K-chunks of 32)
    FragBF qf[4];
    {
        const bf16* qrow = qbase + (size_t)(qi0 + l16) * q_row_stride;
#pragma unroll
        for (int kc = 0; kc < 4; ++kc) {
#pragma unroll
            for (int j = 0; j < 8; ++j) {
                const int koff = kc * 32 + 2 * (j & 3) + ((j >= 4) ? 16 : 0) + hi * 8;
                qf[kc].u[j] = *(const unsigned int*)(qrow + koff);
            }
        }
    }

    floatx8 o[8] = {};                 // 16 x 128 output accumulator (f32)
    float m_i = -1e30f, l_i = 0.f;     // per-row state, valid in lanes 0..15

    const int kend = causal ? (qi0 + 16) : T;
    for (int kj0 = 0; kj0 < kend; kj0 += 32) {
        // ---- S = (Q @ K^T): two 16x16 column blocks over d=128 ----
        floatx8 s0 = {}, s1 = {};
#pragma unroll
        for (int g = 0; g < 2; ++g) {
            const bf16* krow =
                kbase + (size_t)(kj0 + g * 16 + l16) * kv_row_stride;
            floatx8 sacc = g ? s1 : s0;
#pragma unroll
            for (int kc = 0; kc < 4; ++kc) {
                FragBF kf;
#pragma unroll
                for (int v = 0; v < 8; ++v)
                    kf.u[v] = *(const unsigned int*)(krow + kc * 32 + 2 * (v & 3) + ((v >= 4) ? 16 : 0) + hi * 8);
                sacc = __builtin_amdgcn_wmma_f32_16x16x32_bf16(
                    false, qf[kc].v, false, kf.v, (short)0, sacc, false, false);
            }
            if (g) s1 = sacc; else s0 = sacc;
        }
        // spill S to LDS in (row, key) order
#pragma unroll
        for (int r = 0; r < 8; ++r) {
            Ssm[r + hi * 8][l16]      = s0[r];
            Ssm[r + hi * 8][16 + l16] = s1[r];
        }
        __syncthreads();

        // ---- online softmax: lane i owns row i (i < 16) ----
        if (lane < 16) {
            const int row = lane;
            const int qpos = qi0 + row;
            float sv[32];
            float mx = -1e30f;
#pragma unroll
            for (int j = 0; j < 32; ++j) {
                float s = Ssm[row][j] * scale;
                if (causal && (kj0 + j > qpos)) s = -1e30f;
                sv[j] = s;
                mx = fmaxf(mx, s);
            }
            const float m_new = fmaxf(m_i, mx);
            const float corr  = __expf(m_i - m_new);
            float ps = 0.f;
#pragma unroll
            for (int j = 0; j < 32; ++j) {
                const float p = __expf(sv[j] - m_new);
                ps += p;
                Psm[row][j] = f2bf(p);
            }
            l_i = l_i * corr + ps;
            m_i = m_new;
            csm[row] = corr;
        }
        __syncthreads();

        // ---- rescale accumulator by per-row correction ----
        float cr[8];
#pragma unroll
        for (int r = 0; r < 8; ++r) cr[r] = csm[r + hi * 8];
#pragma unroll
        for (int f = 0; f < 8; ++f)
#pragma unroll
            for (int r = 0; r < 8; ++r) o[f][r] *= cr[r];

        // ---- O += P (16x32) @ V (32x128) ----
        FragBF pf;
#pragma unroll
        for (int j = 0; j < 8; ++j) {
            const int koff = 2 * (j & 3) + ((j >= 4) ? 16 : 0) + hi * 8;
            pf.u[j] = *(const unsigned int*)&Psm[l16][koff];
        }
#pragma unroll
        for (int f = 0; f < 8; ++f) {
            FragBF vf;
#pragma unroll
            for (int v = 0; v < 8; ++v) {
                const int key0 = kj0 + 2 * (v & 3) + ((v >= 4) ? 16 : 0) + hi * 8;
                const bf16* v0 =
                    vbase + (size_t)key0 * kv_row_stride + f * 16 + l16;
                const unsigned short a = *(const unsigned short*)v0;
                const unsigned short c = *(const unsigned short*)(v0 + kv_row_stride);
                vf.u[v] = (unsigned int)a | ((unsigned int)c << 16);
            }
            o[f] = __builtin_amdgcn_wmma_f32_16x16x32_bf16(
                false, pf.v, false, vf.v, (short)0, o[f], false, false);
        }
        __syncthreads();
    }

    // ---- finalize: divide by row sums, store bf16 ----
    if (lane < 16) lsm[lane] = l_i;
    __syncthreads();
    float invl[8];
#pragma unroll
    for (int r = 0; r < 8; ++r) invl[r] = __builtin_amdgcn_rcpf(lsm[r + hi * 8]);
    bf16* obase = O + (size_t)b * T * o_row_stride + (size_t)h * D_HEAD;
#pragma unroll
    for (int f = 0; f < 8; ++f) {
#pragma unroll
        for (int r = 0; r < 8; ++r) {
            const int row = qi0 + r + hi * 8;
            obase[(size_t)row * o_row_stride + f * 16 + l16] =
                f2bf(o[f][r] * invl[r]);
        }
    }
}

// ---------------------------------------------------------------------------
// Host-side orchestration
// ---------------------------------------------------------------------------
extern "C" void kernel_launch(void* const* d_in, const int* in_sizes, int n_in,
                              void* d_out, int out_size, void* d_ws, size_t ws_size,
                              hipStream_t stream) {
    (void)in_sizes; (void)n_in; (void)out_size; (void)ws_size;
    const float* x          = (const float*)d_in[0];
    const float* k_ctx      = (const float*)d_in[1];
    const float* v_ctx      = (const float*)d_in[2];
    const float* wq         = (const float*)d_in[3];
    const float* wo_cross   = (const float*)d_in[4];
    const float* norm_q_w   = (const float*)d_in[5];
    const float* wqkv       = (const float*)d_in[6];
    const float* wo_self    = (const float*)d_in[7];
    const float* norm_qkv_w = (const float*)d_in[8];
    const float* ffn_w1     = (const float*)d_in[9];
    const float* ffn_w2     = (const float*)d_in[10];
    const float* ffn_norm_w = (const float*)d_in[11];
    float* out = (float*)d_out;

    const long MT = (long)BATCH * SEQ;          // 4096 rows
    char* ws = (char*)d_ws;
    size_t off = 0;
    auto alloc = [&](size_t bytes) -> char* {
        char* p = ws + off;
        off += (bytes + 255) & ~(size_t)255;
        return p;
    };
    bf16* wq_bf   = (bf16*)alloc((size_t)D_MODEL * D_MODEL * 2);
    bf16* woc_bf  = (bf16*)alloc((size_t)D_MODEL * D_MODEL * 2);
    bf16* wqkv_bf = (bf16*)alloc((size_t)D_MODEL * 3 * D_MODEL * 2);
    bf16* wos_bf  = (bf16*)alloc((size_t)D_MODEL * D_MODEL * 2);
    bf16* w1_bf   = (bf16*)alloc((size_t)D_MODEL * D_FF * 2);
    bf16* w2_bf   = (bf16*)alloc((size_t)D_FF * D_MODEL * 2);
    bf16* kctx_bf = (bf16*)alloc((size_t)MT * GKV * D_HEAD * 2);
    bf16* vctx_bf = (bf16*)alloc((size_t)MT * GKV * D_HEAD * 2);
    bf16* xn_bf   = (bf16*)alloc((size_t)MT * D_MODEL * 2);
    bf16* q_bf    = (bf16*)alloc((size_t)MT * D_MODEL * 2);
    bf16* qkv_bf  = (bf16*)alloc((size_t)MT * 3 * D_MODEL * 2);
    bf16* ctx_bf  = (bf16*)alloc((size_t)MT * D_MODEL * 2);
    float* x1     = (float*)alloc((size_t)MT * D_MODEL * 4);
    bf16* h_bf    = (bf16*)alloc((size_t)MT * D_FF * 2);

    const float scale = 0.08838834764831845f;   // 1/sqrt(128)
    const dim3 blk256(256), blk32(32);

    // --- weight / kv conversions to bf16 ---
    k_f32_to_bf16<<<4096, blk256, 0, stream>>>(wq,      wq_bf,   (long)D_MODEL * D_MODEL);
    k_f32_to_bf16<<<4096, blk256, 0, stream>>>(wo_cross,woc_bf,  (long)D_MODEL * D_MODEL);
    k_f32_to_bf16<<<4096, blk256, 0, stream>>>(wqkv,    wqkv_bf, (long)D_MODEL * 3 * D_MODEL);
    k_f32_to_bf16<<<4096, blk256, 0, stream>>>(wo_self, wos_bf,  (long)D_MODEL * D_MODEL);
    k_f32_to_bf16<<<4096, blk256, 0, stream>>>(ffn_w1,  w1_bf,   (long)D_MODEL * D_FF);
    k_f32_to_bf16<<<4096, blk256, 0, stream>>>(ffn_w2,  w2_bf,   (long)D_FF * D_MODEL);
    k_f32_to_bf16<<<4096, blk256, 0, stream>>>(k_ctx,   kctx_bf, MT * GKV * D_HEAD);
    k_f32_to_bf16<<<4096, blk256, 0, stream>>>(v_ctx,   vctx_bf, MT * GKV * D_HEAD);

    // --- cross attention: q = rmsnorm(x) @ wq ---
    k_rmsnorm_bf16<<<MT, blk256, 0, stream>>>(x, norm_q_w, xn_bf);
    k_gemm_bf16_wmma<false, false, true>
        <<<dim3(D_MODEL / 128, MT / 128), blk256, 0, stream>>>(
        xn_bf, wq_bf, nullptr, nullptr, q_bf, (int)MT, D_MODEL, D_MODEL);
    k_flash_attn<<<dim3(SEQ / 16, N_HEADS, BATCH), blk32, 0, stream>>>(
        q_bf, (long)D_MODEL, kctx_bf, vctx_bf, (long)(GKV * D_HEAD), 4,
        ctx_bf, (long)D_MODEL, SEQ, 0, scale);
    // x1 = x + ctx @ wo_cross
    k_gemm_bf16_wmma<true, false, false>
        <<<dim3(D_MODEL / 128, MT / 128), blk256, 0, stream>>>(
        ctx_bf, woc_bf, x, x1, nullptr, (int)MT, D_MODEL, D_MODEL);

    // --- causal self attention ---
    k_rmsnorm_bf16<<<MT, blk256, 0, stream>>>(x1, norm_qkv_w, xn_bf);
    k_gemm_bf16_wmma<false, false, true>
        <<<dim3(3 * D_MODEL / 128, MT / 128), blk256, 0, stream>>>(
        xn_bf, wqkv_bf, nullptr, nullptr, qkv_bf, (int)MT, 3 * D_MODEL, D_MODEL);
    k_flash_attn<<<dim3(SEQ / 16, N_HEADS, BATCH), blk32, 0, stream>>>(
        qkv_bf, (long)(3 * D_MODEL),
        qkv_bf + D_MODEL, qkv_bf + 2 * D_MODEL, (long)(3 * D_MODEL), 1,
        ctx_bf, (long)D_MODEL, SEQ, 1, scale);
    // x2 = x1 + ctx2 @ wo_self   (written to d_out)
    k_gemm_bf16_wmma<true, false, false>
        <<<dim3(D_MODEL / 128, MT / 128), blk256, 0, stream>>>(
        ctx_bf, wos_bf, x1, out, nullptr, (int)MT, D_MODEL, D_MODEL);

    // --- FFN: out = x2 + silu(rmsnorm(x2) @ w1) @ w2 ---
    k_rmsnorm_bf16<<<MT, blk256, 0, stream>>>(out, ffn_norm_w, xn_bf);
    k_gemm_bf16_wmma<false, true, true>
        <<<dim3(D_FF / 128, MT / 128), blk256, 0, stream>>>(
        xn_bf, w1_bf, nullptr, nullptr, h_bf, (int)MT, D_FF, D_MODEL);
    k_gemm_bf16_wmma<true, false, false>
        <<<dim3(D_MODEL / 128, MT / 128), blk256, 0, stream>>>(
        h_bf, w2_bf, out, out, nullptr, (int)MT, D_MODEL, D_FF);
}
